// LSTMAttentionDot_47227460387524
// MI455X (gfx1250) — matrix operand order repaired and checked by reading, not compiled, not run on hardware
//
#include <hip/hip_runtime.h>

// ---------------- problem constants (from reference) ----------------
#define BATCH   64
#define SEQ     512
#define DIN     1024
#define DHID    1024
#define NSTEPS  64
#define NGATES  4096   // 4*DHID

typedef unsigned short ushort_t;
typedef unsigned int   uint_t;

typedef __attribute__((ext_vector_type(16))) __bf16 v16bf;
typedef __attribute__((ext_vector_type(8)))  float  v8f;
typedef __attribute__((ext_vector_type(4)))  unsigned int v4u;
typedef __attribute__((ext_vector_type(8)))  int v8i;
typedef __attribute__((ext_vector_type(4)))  int v4i;

union FragBF { v16bf v; uint_t u[8]; };

// ---------------- helpers ----------------
static __device__ __forceinline__ ushort_t f2bf(float x) {
    union { float f; uint_t u; } v; v.f = x;
    uint_t r = v.u + 0x7fffu + ((v.u >> 16) & 1u);   // round-to-nearest-even
    return (ushort_t)(r >> 16);
}
static __device__ __forceinline__ float bf2f(ushort_t h) {
    union { uint_t u; float f; } v; v.u = ((uint_t)h) << 16;
    return v.f;
}
static __device__ __forceinline__ float sigmoidf(float x) {
    return 1.0f / (1.0f + __expf(-x));
}

// ---------------- bf16 WMMA GEMM: C[64,N] = A[64,K] * B[K,N] ----------------
// A: bf16 row-major [64,K]; B: bf16 row-major [K,N] (pre-transposed weights).
// 128 threads = 4 waves; block computes 64x64 tile (wave w -> rows 16w..16w+15).
// A tile is staged to LDS via the Tensor Data Mover (tensor_load_to_lds), B
// tile via vectorized loads (static 8-deep unroll so the loads batch in
// flight) with an [n][k] transposed pack so fragment reads are contiguous
// ds_load_b128's.
__global__ __launch_bounds__(128) void gemm_bf16_wmma(
    const ushort_t* __restrict__ A,
    const ushort_t* __restrict__ B,
    const float*    __restrict__ bias,   // len N or nullptr
    float*    outf, int ldf,             // fp32 out or nullptr
    ushort_t* outbf, int ldbf,           // bf16 out or nullptr
    int K, int N, int do_tanh)
{
    __shared__ uint_t sA[64 * 16];   // [row][kpair] 64x32 bf16 (row-major, TDM dest)
    __shared__ uint_t sB[64 * 16];   // [n][kpair]   64 cols x 32 k (transposed-packed)

    const int nblock = blockIdx.x * 64;
    const int wave = threadIdx.x >> 5;
    const int lane = threadIdx.x & 31;
    const int m0   = wave * 16;

    v8f acc[4];
#pragma unroll
    for (int t = 0; t < 4; ++t)
#pragma unroll
        for (int v = 0; v < 8; ++v) acc[t][v] = 0.0f;

    // ---- TDM descriptor (invariant part): A tile = 64 rows x 32 bf16 ----
    // D# group1: data_size=1(2B) | tensor_dim0=K | tensor_dim1=64 |
    //            tile_dim0=32 | tile_dim1=64 | tensor_dim0_stride=K
    const unsigned long long abase = (unsigned long long)(const void*)A;
    const unsigned int lds_off = (unsigned int)(unsigned long long)(void*)sA;
    v8i g1;
    g1[0] = 0x00010000;                                   // data_size = 2 bytes
    g1[1] = (int)(((unsigned)K & 0xffffu) << 16);         // tensor_dim0 lo16
    g1[2] = (int)((((unsigned)K >> 16) & 0xffffu) | (64u << 16)); // dim0 hi | dim1 lo
    g1[3] = (int)(32u << 16);                             // dim1 hi=0 | tile_dim0=32
    g1[4] = 64;                                           // tile_dim1=64 | tile_dim2=0
    g1[5] = (int)K;                                       // tensor_dim0_stride lo32
    g1[6] = 0;                                            // stride hi | dim1_stride lo
    g1[7] = 0;
    const v4i gz4 = {0, 0, 0, 0};                         // groups 2/3 unused (2D)
    const v8i gz8 = {0, 0, 0, 0, 0, 0, 0, 0};

    // B staging: thread's fixed (k, n-pair) footprint, 4 rows apart per step
    const int bk0 = threadIdx.x >> 5;          // base k row (0..3), step +4 -> only with e=tid+i*128: k = e>>5
    const int bnp = threadIdx.x & 31;          // n-pair (n = 2*bnp, 2*bnp+1)
    const uint_t* Bd = (const uint_t*)B;       // dword view of B
    const int bstride = N >> 1;                // dwords per B row
    const uint_t* Bbase = Bd + (size_t)bk0 * bstride + (nblock >> 1) + bnp;
    ushort_t* sB16 = (ushort_t*)sB;
    ushort_t* sBlo = sB16 + (size_t)(bnp << 1) * 32 + bk0;        // [2*bnp][k]
    ushort_t* sBhi = sBlo + 32;                                   // [2*bnp+1][k]

    for (int kc = 0; kc < K; kc += 32) {
        __syncthreads();    // previous iteration's fragment reads done

        // ---- async A tile: TDM global->LDS (one issue per workgroup) ----
        if (threadIdx.x < 32) {
            unsigned long long ga = abase + (unsigned long long)kc * 2ull;
            v4u g0;
            g0[0] = 1u;                                   // count=1 (valid)
            g0[1] = lds_off;                              // lds_addr
            g0[2] = (unsigned int)(ga & 0xffffffffu);     // global_addr lo
            g0[3] = (unsigned int)((ga >> 32) & 0x1ffffffu) | (2u << 30); // hi | type=2
            __builtin_amdgcn_tensor_load_to_lds(g0, g1, gz4, gz4, gz8, 0);
        }

        // prefetch next K-slab of B into L2 (global_prefetch_b8) before stores
        if (kc + 32 < K) {
            __builtin_prefetch(&B[(size_t)(kc + 32 + (threadIdx.x >> 2)) * N + nblock + ((threadIdx.x & 3) << 4)], 0, 1);
        }

        // ---- stage B tile transposed: sB16[n*32+k] = B[kc+k][nblock+n] ----
        // static trip count: 8 k-rows per thread (rows bk0, bk0+4, ..., bk0+28)
        {
            uint_t dw[8];
            const uint_t* p = Bbase + (size_t)kc * bstride;
#pragma unroll
            for (int i = 0; i < 8; ++i)
                dw[i] = p[(size_t)(i * 4) * bstride];     // 8 loads in flight
#pragma unroll
            for (int i = 0; i < 8; ++i) {
                int k = bk0 + i * 4;
                sBlo[k] = (ushort_t)(dw[i] & 0xffffu);
                sBhi[k] = (ushort_t)(dw[i] >> 16);
            }
        }

        if (threadIdx.x < 32) __builtin_amdgcn_s_wait_tensorcnt(0);
        __syncthreads();

        // ---- A fragment (16-bit A 16x32 lane layout) ----
        FragBF a;
        {
            int row = m0 + (lane & 15);
            int kh4 = (lane >> 4) * 8;      // kh*16 bf16 = kh*8 dwords
#pragma unroll
            for (int v = 0; v < 4; ++v) a.u[v]     = sA[row * 16 + kh4 + v];
#pragma unroll
            for (int v = 0; v < 4; ++v) a.u[4 + v] = sA[row * 16 + 4 + kh4 + v];
        }
        // ---- load all 4 B fragments, then issue 4 WMMAs back-to-back ----
        FragBF bfr[4];
#pragma unroll
        for (int t = 0; t < 4; ++t) {
            int n  = t * 16 + (lane & 15);
            int kb = (lane >> 4) * 8;       // kb*16 bf16 = 8 dwords
#pragma unroll
            for (int v = 0; v < 8; ++v) bfr[t].u[v] = sB[n * 16 + kb + v];
        }
#pragma unroll
        for (int t = 0; t < 4; ++t) {
            acc[t] = __builtin_amdgcn_wmma_f32_16x16x32_bf16(
                false, a.v, false, bfr[t].v, (short)0, acc[t], false, false);
        }
    }

    // ---- epilogue ----
#pragma unroll
    for (int t = 0; t < 4; ++t) {
        int n = nblock + t * 16 + (lane & 15);
#pragma unroll
        for (int v = 0; v < 8; ++v) {
            int m = m0 + v + (lane >> 4) * 8;
            float val = acc[t][v];
            if (bias)    val += bias[n];
            if (do_tanh) val = tanhf(val);
            if (outf)  outf[(size_t)m * ldf + n]  = val;
            if (outbf) outbf[(size_t)m * ldbf + n] = f2bf(val);
        }
    }
}

// ---------------- weight transpose + bf16 convert: WT[c*R+r] = bf(W[r*C+c]) ----
__global__ void transpose_to_bf16(const float* __restrict__ W, ushort_t* __restrict__ WT,
                                  int R, int C)
{
    size_t idx = (size_t)blockIdx.x * 256 + threadIdx.x;
    if (idx < (size_t)R * C) {
        int r = (int)(idx / C), c = (int)(idx % C);
        WT[(size_t)c * R + r] = f2bf(W[idx]);
    }
}

// ---------------- init state: h->bf16, c copy ----------------
__global__ void init_state(const float* __restrict__ hx, const float* __restrict__ cx,
                           ushort_t* __restrict__ hbf, float* __restrict__ c)
{
    int idx = blockIdx.x * 256 + threadIdx.x;   // 65536
    hbf[idx] = f2bf(hx[idx]);
    c[idx]   = cx[idx];
}

// ---------------- LSTM pointwise ----------------
__global__ void lstm_pointwise(const float* __restrict__ gates, float* __restrict__ c,
                               ushort_t* __restrict__ hy_bf)
{
    int idx = blockIdx.x * 256 + threadIdx.x;   // 65536
    int b = idx >> 10, d = idx & 1023;
    const float* g = gates + (size_t)b * NGATES;
    float gi = sigmoidf(g[d]);
    float gf = sigmoidf(g[1024 + d]);
    float gg = tanhf(g[2048 + d]);
    float go = sigmoidf(g[3072 + d]);
    float cy = gf * c[idx] + gi * gg;
    c[idx] = cy;
    hy_bf[idx] = f2bf(go * tanhf(cy));
}

// ---------------- attention scores: one wave per (b,s) row ----------------
__global__ void attn_scores(const float* __restrict__ ctx, const ushort_t* __restrict__ cat,
                            const float* __restrict__ mask, float* __restrict__ attn)
{
    int b = blockIdx.y;
    int s = blockIdx.x * 8 + (threadIdx.x >> 5);
    int lane = threadIdx.x & 31;
    const float* crow = ctx + ((size_t)b * SEQ + s) * DIN;
    const ushort_t* inp = cat + (size_t)b * 2048 + 1024;
    float acc = 0.0f;
    for (int d = lane; d < DIN; d += 32)
        acc += crow[d] * bf2f(inp[d]);
#pragma unroll
    for (int off = 16; off > 0; off >>= 1)
        acc += __shfl_xor(acc, off, 32);
    if (lane == 0) {
        int i = b * SEQ + s;
        attn[i] = acc - (1.0f - mask[i]) * 100000.0f;
    }
}

// ---------------- softmax over S=512, one block per batch row ----------------
__global__ void softmax512(float* __restrict__ attn)
{
    __shared__ float red[256];
    float* row = attn + (size_t)blockIdx.x * SEQ;
    int t = threadIdx.x;
    float a0 = row[t], a1 = row[t + 256];
    red[t] = fmaxf(a0, a1);
    __syncthreads();
    for (int o = 128; o > 0; o >>= 1) {
        if (t < o) red[t] = fmaxf(red[t], red[t + o]);
        __syncthreads();
    }
    float mx = red[0];
    __syncthreads();
    float e0 = __expf(a0 - mx), e1 = __expf(a1 - mx);
    red[t] = e0 + e1;
    __syncthreads();
    for (int o = 128; o > 0; o >>= 1) {
        if (t < o) red[t] += red[t + o];
        __syncthreads();
    }
    float inv = 1.0f / red[0];
    row[t]       = e0 * inv;
    row[t + 256] = e1 * inv;
}

// ---------------- weighted ctx: cat[b,0:1024] = bf(sum_s attn * ctx) ----------
__global__ void weighted_ctx(const float* __restrict__ attn, const float* __restrict__ ctx,
                             ushort_t* __restrict__ cat)
{
    int b = blockIdx.y;
    int d = blockIdx.x * 256 + threadIdx.x;
    const float* c = ctx + (size_t)b * SEQ * DIN + d;
    const float* a = attn + (size_t)b * SEQ;
    float acc = 0.0f;
    for (int s = 0; s < SEQ; ++s)
        acc += a[s] * c[(size_t)s * DIN];
    cat[(size_t)b * 2048 + d] = f2bf(acc);
}

// ---------------- finalize: hx/cx tails of d_out ----------------
__global__ void finalize(const float* __restrict__ outbase, const float* __restrict__ c,
                         float* __restrict__ hx_out, float* __restrict__ cx_out)
{
    int idx = blockIdx.x * 256 + threadIdx.x;   // 65536
    int b = idx >> 10, d = idx & 1023;
    hx_out[idx] = outbase[((size_t)b * NSTEPS + (NSTEPS - 1)) * DHID + d];
    cx_out[idx] = c[idx];
}

// ---------------- launch ----------------
extern "C" void kernel_launch(void* const* d_in, const int* in_sizes, int n_in,
                              void* d_out, int out_size, void* d_ws, size_t ws_size,
                              hipStream_t stream)
{
    const float* hx   = (const float*)d_in[0];
    const float* cx   = (const float*)d_in[1];
    const float* ctx  = (const float*)d_in[2];
    const float* mask = (const float*)d_in[3];
    const float* Wh   = (const float*)d_in[4];
    const float* bh   = (const float*)d_in[5];
    const float* Win  = (const float*)d_in[6];
    const float* Wout = (const float*)d_in[7];

    float* out = (float*)d_out;   // [B, STEPS, DHID] + hx[B,DHID] + cx[B,DHID]

    // workspace layout (bytes)
    char* ws = (char*)d_ws;
    size_t off = 0;
    ushort_t* WhT   = (ushort_t*)(ws + off); off += (size_t)DHID * NGATES * 2;        // [1024,4096]
    ushort_t* WinT  = (ushort_t*)(ws + off); off += (size_t)DHID * DIN * 2;           // [1024,1024]
    ushort_t* WoutT = (ushort_t*)(ws + off); off += (size_t)2 * DIN * DHID * 2;       // [2048,1024]
    ushort_t* h_bf  = (ushort_t*)(ws + off); off += (size_t)BATCH * DHID * 2;
    ushort_t* hy_bf = (ushort_t*)(ws + off); off += (size_t)BATCH * DHID * 2;
    float*    c_st  = (float*)   (ws + off); off += (size_t)BATCH * DHID * 4;
    float*    gates = (float*)   (ws + off); off += (size_t)BATCH * NGATES * 4;
    ushort_t* cat   = (ushort_t*)(ws + off); off += (size_t)BATCH * 2048 * 2;
    float*    attn  = (float*)   (ws + off); off += (size_t)BATCH * SEQ * 4;
    (void)ws_size; (void)in_sizes; (void)n_in; (void)out_size;

    // one-time (per launch) conversions
    transpose_to_bf16<<<(NGATES * DHID + 255) / 256, 256, 0, stream>>>(Wh, WhT, NGATES, DHID);
    transpose_to_bf16<<<(DIN * DHID + 255) / 256, 256, 0, stream>>>(Win, WinT, DIN, DHID);
    transpose_to_bf16<<<(DHID * 2 * DIN + 255) / 256, 256, 0, stream>>>(Wout, WoutT, DHID, 2 * DIN);
    init_state<<<BATCH * DHID / 256, 256, 0, stream>>>(hx, cx, h_bf, c_st);

    for (int t = 0; t < NSTEPS; ++t) {
        // gates = h @ Wh^T + bh   (M=64, K=1024, N=4096)
        gemm_bf16_wmma<<<NGATES / 64, 128, 0, stream>>>(
            h_bf, WhT, bh, gates, NGATES, (ushort_t*)nullptr, 0, DHID, NGATES, 0);
        // LSTM pointwise -> c, hy(bf16)
        lstm_pointwise<<<BATCH * DHID / 256, 256, 0, stream>>>(gates, c_st, hy_bf);
        // inp = hy @ W_in^T  (M=64, K=1024, N=1024) -> bf16 into cat[:,1024:2048]
        gemm_bf16_wmma<<<DIN / 64, 128, 0, stream>>>(
            hy_bf, WinT, (const float*)nullptr, (float*)nullptr, 0,
            cat + 1024, 2048, DHID, DIN, 0);
        // attention
        attn_scores<<<dim3(SEQ / 8, BATCH), 256, 0, stream>>>(ctx, cat, mask, attn);
        softmax512<<<BATCH, 256, 0, stream>>>(attn);
        weighted_ctx<<<dim3(DIN / 256, BATCH), 256, 0, stream>>>(attn, ctx, cat);
        // h_tilde = tanh(cat @ W_out^T)  (M=64, K=2048, N=1024)
        //   fp32 -> out[:, t, :] (row stride STEPS*DHID), bf16 -> h_bf for next step
        gemm_bf16_wmma<<<DHID / 64, 128, 0, stream>>>(
            cat, WoutT, (const float*)nullptr,
            out + (size_t)t * DHID, NSTEPS * DHID,
            h_bf, DHID, 2 * DIN, DHID, 1);
    }

    float* hx_out = out + (size_t)BATCH * NSTEPS * DHID;
    float* cx_out = hx_out + (size_t)BATCH * DHID;
    finalize<<<BATCH * DHID / 256, 256, 0, stream>>>(out, c_st, hx_out, cx_out);
}
